// PreGnn_3169685864863
// MI455X (gfx1250) — matrix operand order repaired and checked by reading, not compiled, not run on hardware
//
#include <hip/hip_runtime.h>
#include <hip/hip_bf16.h>
#include <math.h>

// ---------------------------------------------------------------------------
// CDNA5 (gfx1250) implementation of the GNN TransformerConv pipeline.
// Matrix math uses v_wmma_f32_16x16x32_f16 (f16 inputs, f32 accumulate).
// Wave32 everywhere; fragment layouts follow CDNA5 ISA.
// LDS tiles use a 72-halfword row stride: 16-B aligned rows for ds_load_b128
// and 36-bank stepping so 16 lanes cover 64 banks conflict-free.
// Bias vectors are folded into the WMMA C operand (accumulator init), which
// removes the WMMA->VALU RAW hazard NOP sequence after each tile.
// ---------------------------------------------------------------------------

typedef _Float16 v16h __attribute__((ext_vector_type(16)));
typedef _Float16 v8h  __attribute__((ext_vector_type(8)));
typedef float    v8f  __attribute__((ext_vector_type(8)));

#define HID 64          // hidden_channels (fixed by reference)
#define LSTR 72         // padded LDS row stride (halfwords)

__device__ inline v8f wmma_f16(v16h a, v16h b, v8f c) {
  // D = A(16x32 f16) * B(32x16 f16) + C(16x16 f32)
  return __builtin_amdgcn_wmma_f32_16x16x32_f16(
      /*neg_a=*/false, a, /*neg_b=*/false, b,
      /*c_mod=*/(short)0, c, /*reuse_a=*/false, /*reuse_b=*/false);
}

__device__ inline v16h cat8(v8h lo, v8h hi) {
  return __builtin_shufflevector(lo, hi, 0, 1, 2, 3, 4, 5, 6, 7,
                                 8, 9, 10, 11, 12, 13, 14, 15);
}

__device__ inline v8f splat8(float f) {
  v8f c;
#pragma unroll
  for (int r = 0; r < 8; ++r) c[r] = f;
  return c;
}

// A fragment from a row-major 16 x HID LDS tile (stride LSTR).
// Lanes 0-15 (row M=lane):  elems 0..7 -> K=kbase+0..7, 8..15 -> kbase+16..23
// Lanes 16-31 (row M=lane-16): elems 0..7 -> kbase+8..15, 8..15 -> kbase+24..31
// Both halves are 8 contiguous halfwords -> two ds_load_b128.
__device__ inline v16h load_A_lds(const _Float16* t, int lane, int kbase) {
  const int m  = lane & 15;
  const int ko = kbase + ((lane < 16) ? 0 : 8);
  v8h lo = *(const v8h*)(t + m * LSTR + ko);
  v8h hi = *(const v8h*)(t + m * LSTR + ko + 16);
  return cat8(lo, hi);
}

// B fragment from a TRANSPOSED [n][k] LDS tile (stride LSTR).
// Lane L holds column N=n0+L%16; elems 0..7 -> K = kbase + (L<16?0:8) + e,
// elems 8..15 -> K = kbase + 16 + (L<16?0:8) + (e-8) -> two ds_load_b128.
__device__ inline v16h load_B_ldsT(const _Float16* wt, int lane, int kbase, int n0) {
  const int n  = n0 + (lane & 15);
  const int ko = kbase + ((lane < 16) ? 0 : 8);
  v8h lo = *(const v8h*)(wt + n * LSTR + ko);
  v8h hi = *(const v8h*)(wt + n * LSTR + ko + 16);
  return cat8(lo, hi);
}

// order-preserving float <-> uint key for atomicMax-based segment max
__device__ inline unsigned fkey(float f) {
  unsigned u = __float_as_uint(f);
  return (u >> 31) ? ~u : (u | 0x80000000u);
}
__device__ inline float fdec(unsigned k) {
  return (k >> 31) ? __uint_as_float(k & 0x7fffffffu) : __uint_as_float(~k);
}

// stage a HIDxHID f32 weight matrix into LDS as transposed f16 [n][k]
__device__ inline void stage_wT(_Float16* wt, const float* __restrict__ W, int tid) {
  for (int i = tid; i < HID * HID; i += 256) {
    const int kk = i >> 6, n = i & 63;
    wt[n * LSTR + kk] = (_Float16)W[i];   // coalesced global read
  }
}

// one projection: out[tile rows] = A @ W + b  (W transposed in LDS;
// bias enters through the accumulator -> no post-WMMA VALU)
__device__ inline void do_proj(const _Float16* wT, const float* __restrict__ bias,
                               float* __restrict__ outbuf, v16h a0, v16h a1,
                               int lane, int tile) {
  const int mb = (lane < 16) ? 0 : 8;
#pragma unroll
  for (int nt = 0; nt < 4; ++nt) {
    const int n = nt * 16 + (lane & 15);
    v8f acc = splat8(bias[n]);           // C = bias (same column per lane)
    acc = wmma_f16(a0, load_B_ldsT(wT, lane, 0, nt * 16), acc);
    acc = wmma_f16(a1, load_B_ldsT(wT, lane, 32, nt * 16), acc);
#pragma unroll
    for (int r = 0; r < 8; ++r)
      outbuf[(size_t)(tile * 16 + mb + r) * HID + n] = acc[r];
  }
}

// ---------------------------------------------------------------------------
// Kernel 1: fused projections  q,k,v = h@W*+b*,  s = h@Ws+bs   (h = emb[x])
// One 16-node tile per wave, 8 waves/block; weights in LDS (transposed f16).
// ---------------------------------------------------------------------------
__global__ __launch_bounds__(256) void k_proj(
    const int* __restrict__ x, const float* __restrict__ emb,
    const float* __restrict__ Wq, const float* __restrict__ bq,
    const float* __restrict__ Wk, const float* __restrict__ bk,
    const float* __restrict__ Wv, const float* __restrict__ bv,
    const float* __restrict__ Ws, const float* __restrict__ bs,
    float* __restrict__ q, float* __restrict__ k, float* __restrict__ v,
    float* __restrict__ s, int nNodes) {
  __shared__ __align__(16) _Float16 wq[HID * LSTR];   // 9 KB each
  __shared__ __align__(16) _Float16 wk[HID * LSTR];
  __shared__ __align__(16) _Float16 wv[HID * LSTR];
  __shared__ __align__(16) _Float16 wsk[HID * LSTR];
  __shared__ __align__(16) _Float16 hA[8][16 * LSTR]; // 2.25 KB per wave

  const int tid  = threadIdx.x;
  const int wave = tid >> 5;
  const int lane = tid & 31;

  stage_wT(wq, Wq, tid);
  stage_wT(wk, Wk, tid);
  stage_wT(wv, Wv, tid);
  stage_wT(wsk, Ws, tid);

  const int nTiles = (nNodes + 15) >> 4;
  int tile = blockIdx.x * 8 + wave;
  if (tile >= nTiles) tile = nTiles - 1;  // duplicate work, identical stores

  // gather h tile = emb[x[node]] as f16
  for (int i = lane; i < 16 * HID; i += 32) {
    const int r = i >> 6, c = i & 63;
    int node = tile * 16 + r;
    if (node >= nNodes) node = nNodes - 1;
    hA[wave][r * LSTR + c] = (_Float16)emb[(size_t)x[node] * HID + c];
  }
  __syncthreads();

  const v16h a0 = load_A_lds(hA[wave], lane, 0);
  const v16h a1 = load_A_lds(hA[wave], lane, 32);

  do_proj(wq,  bq, q, a0, a1, lane, tile);
  do_proj(wk,  bk, k, a0, a1, lane, tile);
  do_proj(wv,  bv, v, a0, a1, lane, tile);
  do_proj(wsk, bs, s, a0, a1, lane, tile);
}

// ---------------------------------------------------------------------------
// Kernel 2: per-edge logits + segment max over dst
// ---------------------------------------------------------------------------
__global__ __launch_bounds__(256) void k_logits(
    const int* __restrict__ src, const int* __restrict__ dst,
    const float* __restrict__ q, const float* __restrict__ kb,
    float* __restrict__ elog, unsigned* __restrict__ mkey, int E) {
  const int e = blockIdx.x * 256 + threadIdx.x;
  if (e >= E) return;
  const int s = src[e], d = dst[e];
  const float4* qa = (const float4*)(q + (size_t)d * HID);
  const float4* ka = (const float4*)(kb + (size_t)s * HID);
  float acc = 0.f;
#pragma unroll
  for (int j = 0; j < HID / 4; ++j) {
    float4 a = qa[j], b = ka[j];
    acc += a.x * b.x + a.y * b.y + a.z * b.z + a.w * b.w;
  }
  acc *= 0.125f;  // 1/sqrt(64)
  elog[e] = acc;
  atomicMax(&mkey[d], fkey(acc));
}

// Kernel 3: e = exp(logit - max[dst]);  denom[dst] += e
__global__ __launch_bounds__(256) void k_expdenom(
    const int* __restrict__ dst, float* __restrict__ elog,
    const unsigned* __restrict__ mkey, float* __restrict__ denom, int E) {
  const int e = blockIdx.x * 256 + threadIdx.x;
  if (e >= E) return;
  const int d = dst[e];
  const float ev = __expf(elog[e] - fdec(mkey[d]));
  elog[e] = ev;
  atomicAdd(&denom[d], ev);
}

// ---------------------------------------------------------------------------
// Kernel 4: agg[dst] += alpha * v[src].  ONE WAVE PER EDGE: edge index is
// wave-uniform so src/dst/alpha scalarize to SMEM; each lane covers 2 columns
// with one global_load_b64 + two no-return f32 atomics.
// ---------------------------------------------------------------------------
__global__ __launch_bounds__(256) void k_agg(
    const int* __restrict__ src, const int* __restrict__ dst,
    const float* __restrict__ elog, const float* __restrict__ denom,
    const float* __restrict__ v, float* __restrict__ agg, int E) {
  const int e = blockIdx.x * 8 + (threadIdx.x >> 5);
  if (e >= E) return;
  const int lane = threadIdx.x & 31;
  const int s = src[e], d = dst[e];
  const float alpha = elog[e] / denom[d];
  const float2 vv = *(const float2*)(v + (size_t)s * HID + lane * 2);
  float* ap = agg + (size_t)d * HID + lane * 2;
  atomicAdd(ap + 0, alpha * vv.x);
  atomicAdd(ap + 1, alpha * vv.y);
}

// Kernel 5: neigh[src] += (agg[dst] + skip[dst]); same wave-per-edge shape.
__global__ __launch_bounds__(256) void k_neigh(
    const int* __restrict__ src, const int* __restrict__ dst,
    const float* __restrict__ agg, const float* __restrict__ skp,
    float* __restrict__ neigh, int E) {
  const int e = blockIdx.x * 8 + (threadIdx.x >> 5);
  if (e >= E) return;
  const int lane = threadIdx.x & 31;
  const int s = src[e], d = dst[e];
  const float2 a  = *(const float2*)(agg + (size_t)d * HID + lane * 2);
  const float2 sk = *(const float2*)(skp + (size_t)d * HID + lane * 2);
  float* np = neigh + (size_t)s * HID + lane * 2;
  atomicAdd(np + 0, a.x + sk.x);
  atomicAdd(np + 1, a.y + sk.y);
}

// ---------------------------------------------------------------------------
// Kernel 6: logits2 = neigh[y] @ W1 + b1 -> out [B x N]   (WMMA)
// grid = (ceil(Ntiles/8), B/16). Per block: stage A (16x64 f16) and a
// transposed 64x128 W1 chunk in LDS (coalesced float4 reads); each of the 8
// waves computes one 16x16 tile with two WMMAs fed by ds_load_b128.
// ---------------------------------------------------------------------------
__global__ __launch_bounds__(256) void k_gemm_big(
    const float* __restrict__ neigh, const int* __restrict__ y,
    const float* __restrict__ W1, const float* __restrict__ b1,
    float* __restrict__ out, int nBatch, int nN) {
  __shared__ __align__(16) _Float16 A[16 * LSTR];    // 2.25 KB
  __shared__ __align__(16) _Float16 BT[128 * LSTR];  // 18 KB, [n][k] layout

  const int tid  = threadIdx.x;
  const int wave = tid >> 5;
  const int lane = tid & 31;
  const int mtile = blockIdx.y;
  const int n0blk = blockIdx.x * 128;

  // stage A tile: rows = neigh[y[mtile*16 + r]]
  for (int i = tid; i < 16 * HID; i += 256) {
    const int r = i >> 6, c = i & 63;
    int row = mtile * 16 + r;
    if (row >= nBatch) row = nBatch - 1;
    A[r * LSTR + c] = (_Float16)neigh[(size_t)y[row] * HID + c];
  }
  // stage BT: W1 rows k=0..63, cols n0blk..n0blk+127, transposed to [n][k]
  for (int idx = tid * 4; idx < HID * 128; idx += 1024) {
    const int kk = idx >> 7;       // 0..63
    const int c  = idx & 127;      // multiple of 4
    long long col = (long long)n0blk + c;
    if (col > (long long)nN - 4) col = nN - 4;  // clamped dup; dead waves skip
    const float4 t = *(const float4*)(W1 + (size_t)kk * nN + col);
    BT[(c + 0) * LSTR + kk] = (_Float16)t.x;
    BT[(c + 1) * LSTR + kk] = (_Float16)t.y;
    BT[(c + 2) * LSTR + kk] = (_Float16)t.z;
    BT[(c + 3) * LSTR + kk] = (_Float16)t.w;
  }
  __syncthreads();

  const int nTiles = (nN + 15) >> 4;
  const int nt = blockIdx.x * 8 + wave;
  if (nt >= nTiles) return;

  const int n = nt * 16 + (lane & 15);
  if (n >= nN) return;

  v8f acc = splat8(b1[n]);  // bias folded into C accumulator
  const v16h a0 = load_A_lds(A, lane, 0);
  const v16h a1 = load_A_lds(A, lane, 32);
  acc = wmma_f16(a0, load_B_ldsT(BT, lane, 0, wave * 16), acc);
  acc = wmma_f16(a1, load_B_ldsT(BT, lane, 32, wave * 16), acc);

  const int mb = (lane < 16) ? 0 : 8;
#pragma unroll
  for (int r = 0; r < 8; ++r) {
    const int row = mtile * 16 + mb + r;
    if (row < nBatch) out[(size_t)row * nN + n] = acc[r];
  }
}

// ---------------------------------------------------------------------------
// Softmax over N per row: max pass, sum-of-exp pass, normalize pass.
// grid = (ceil(N/1024), B), 256 threads, 4 elems/thread (float4).
// ---------------------------------------------------------------------------
__global__ __launch_bounds__(256) void k_rowmax(const float* __restrict__ out,
                                                unsigned* __restrict__ rkey,
                                                int nN) {
  __shared__ float red[256];
  const int row  = blockIdx.y;
  const int base = blockIdx.x * 1024 + threadIdx.x * 4;
  const float* p = out + (size_t)row * nN;
  float m = -INFINITY;
  if (base + 3 < nN) {
    float4 t = *(const float4*)(p + base);
    m = fmaxf(fmaxf(t.x, t.y), fmaxf(t.z, t.w));
  } else {
    for (int i = base; i < nN; ++i) m = fmaxf(m, p[i]);
  }
  red[threadIdx.x] = m;
  __syncthreads();
  for (int off = 128; off; off >>= 1) {
    if (threadIdx.x < off)
      red[threadIdx.x] = fmaxf(red[threadIdx.x], red[threadIdx.x + off]);
    __syncthreads();
  }
  if (threadIdx.x == 0) atomicMax(&rkey[row], fkey(red[0]));
}

__global__ __launch_bounds__(256) void k_rowsum(const float* __restrict__ out,
                                                const unsigned* __restrict__ rkey,
                                                float* __restrict__ rsum, int nN) {
  __shared__ float red[256];
  const int row  = blockIdx.y;
  const int base = blockIdx.x * 1024 + threadIdx.x * 4;
  const float* p = out + (size_t)row * nN;
  const float m  = fdec(rkey[row]);
  float s = 0.f;
  if (base + 3 < nN) {
    float4 t = *(const float4*)(p + base);
    s = __expf(t.x - m) + __expf(t.y - m) + __expf(t.z - m) + __expf(t.w - m);
  } else {
    for (int i = base; i < nN; ++i) s += __expf(p[i] - m);
  }
  red[threadIdx.x] = s;
  __syncthreads();
  for (int off = 128; off; off >>= 1) {
    if (threadIdx.x < off) red[threadIdx.x] += red[threadIdx.x + off];
    __syncthreads();
  }
  if (threadIdx.x == 0) atomicAdd(&rsum[row], red[0]);
}

__global__ __launch_bounds__(256) void k_norm(float* __restrict__ out,
                                              const unsigned* __restrict__ rkey,
                                              const float* __restrict__ rsum,
                                              int nN) {
  const int row  = blockIdx.y;
  const int base = blockIdx.x * 1024 + threadIdx.x * 4;
  float* p = out + (size_t)row * nN;
  const float m   = fdec(rkey[row]);
  const float inv = 1.f / rsum[row];
  if (base + 3 < nN) {
    float4 t = *(const float4*)(p + base);
    t.x = __expf(t.x - m) * inv;
    t.y = __expf(t.y - m) * inv;
    t.z = __expf(t.z - m) * inv;
    t.w = __expf(t.w - m) * inv;
    *(float4*)(p + base) = t;
  } else {
    for (int i = base; i < nN; ++i) p[i] = __expf(p[i] - m) * inv;
  }
}

// ---------------------------------------------------------------------------
extern "C" void kernel_launch(void* const* d_in, const int* in_sizes, int n_in,
                              void* d_out, int out_size, void* d_ws, size_t ws_size,
                              hipStream_t stream) {
  const int*   x    = (const int*)d_in[0];
  const int*   ei   = (const int*)d_in[1];
  const int*   y    = (const int*)d_in[2];
  const float* emb  = (const float*)d_in[3];
  const float* Wq   = (const float*)d_in[4];
  const float* bq   = (const float*)d_in[5];
  const float* Wk   = (const float*)d_in[6];
  const float* bk   = (const float*)d_in[7];
  const float* Wv   = (const float*)d_in[8];
  const float* bv   = (const float*)d_in[9];
  const float* Ws   = (const float*)d_in[10];
  const float* bs   = (const float*)d_in[11];
  const float* W1   = (const float*)d_in[12];
  const float* b1   = (const float*)d_in[13];
  float* out = (float*)d_out;

  const int N = in_sizes[0];
  const int E = in_sizes[1] / 2;
  const int B = in_sizes[2];
  const int* src = ei;
  const int* dst = ei + E;

  // --- workspace carve-up (f32 words) ---
  const size_t NH = (size_t)N * HID;
  float*    q     = (float*)d_ws;
  float*    kbuf  = q + NH;
  float*    vbuf  = kbuf + NH;
  float*    sbuf  = vbuf + NH;
  float*    agg   = sbuf + NH;    // attention aggregate
  float*    neigh = agg + NH;     // second scatter result
  float*    elog  = neigh + NH;   // [E] logits -> exp values
  unsigned* mkey  = (unsigned*)(elog + E);   // [N] segment-max keys
  float*    denom = (float*)(mkey + N);      // [N]
  unsigned* rkey  = (unsigned*)(denom + N);  // [B] row-max keys
  float*    rsum  = (float*)(rkey + B);      // [B]

  // zero the accumulators (capture-legal async memsets)
  hipMemsetAsync(agg, 0, 2 * NH * sizeof(float), stream);            // agg+neigh
  hipMemsetAsync(mkey, 0, (2 * (size_t)N + 2 * (size_t)B) * 4, stream);

  const int nTiles = (N + 15) / 16;
  const int mTiles = (B + 15) / 16;

  // 1) projections
  k_proj<<<(nTiles + 7) / 8, 256, 0, stream>>>(x, emb, Wq, bq, Wk, bk, Wv, bv,
                                               Ws, bs, q, kbuf, vbuf, sbuf, N);
  // 2) edge logits + segment max
  k_logits<<<(E + 255) / 256, 256, 0, stream>>>(src, dst, q, kbuf, elog, mkey, E);
  // 3) exp + denom
  k_expdenom<<<(E + 255) / 256, 256, 0, stream>>>(dst, elog, mkey, denom, E);
  // 4) agg[dst] += alpha * v[src]   (one wave per edge)
  k_agg<<<(E + 7) / 8, 256, 0, stream>>>(src, dst, elog, denom, vbuf, agg, E);
  // 5) neigh[src] += agg[dst] + skip[dst]   (one wave per edge)
  k_neigh<<<(E + 7) / 8, 256, 0, stream>>>(src, dst, agg, sbuf, neigh, E);
  // 6) big GEMM: out = neigh[y] @ W1 + b1
  {
    dim3 grid((nTiles + 7) / 8, mTiles);
    k_gemm_big<<<grid, 256, 0, stream>>>(neigh, y, W1, b1, out, B, N);
  }
  // 7-9) row softmax over N
  {
    dim3 grid((N + 1023) / 1024, B);
    k_rowmax<<<grid, 256, 0, stream>>>(out, rkey, N);
    k_rowsum<<<grid, 256, 0, stream>>>(out, rkey, rsum, N);
    k_norm  <<<grid, 256, 0, stream>>>(out, rkey, rsum, N);
  }
}
